// Seq2Seq_9723805958236
// MI455X (gfx1250) — matrix operand order repaired and checked by reading, not compiled, not run on hardware
//
#include <hip/hip_runtime.h>
#include <hip/hip_bf16.h>
#include <math.h>

#define H   1024
#define ML  128
#define V   32000
#define SOS 0

typedef __attribute__((ext_vector_type(16))) _Float16     v16h;
typedef __attribute__((ext_vector_type(8)))  _Float16     v8h;
typedef __attribute__((ext_vector_type(8)))  float        v8f;
typedef __attribute__((ext_vector_type(4)))  unsigned int u32x4;
typedef __attribute__((ext_vector_type(4)))  int          i32x4;
typedef __attribute__((ext_vector_type(8)))  int          i32x8;

// ---------------------------------------------------------------------------
// CDNA5 async global->LDS staging (ASYNCcnt-tracked), per-lane b128 copies.
// ---------------------------------------------------------------------------
__device__ __forceinline__ void async_b128_to_lds(unsigned lds_byte_addr,
                                                  const void* gaddr)
{
    asm volatile("global_load_async_to_lds_b128 %0, %1, off"
                 :: "v"(lds_byte_addr),
                    "v"((unsigned long long)(uintptr_t)gaddr)
                 : "memory");
}
__device__ __forceinline__ void wait_asynccnt0()
{
    asm volatile("s_wait_asynccnt 0x0" ::: "memory");
}

// ---------------------------------------------------------------------------
// CDNA5 Tensor Data Mover: 1D tile (cols f16 elements) global -> LDS.
// D# built per CDNA5 ISA (count=1, data_size=2B,
// tensor_dim0 = tile_dim0 = cols, tensor_dim1 = tile_dim1 = 1,
// tensor_dim0_stride = cols, type=2).  TENSORcnt-tracked.
// ---------------------------------------------------------------------------
__device__ __forceinline__ void tdm_load_row_to_lds(const void* gaddr,
                                                    unsigned lds_byte_addr,
                                                    int cols)
{
    const unsigned long long ga = (unsigned long long)(uintptr_t)gaddr;
    u32x4 g0;
    g0[0] = 1u;                                        // count=1 (valid), user mode
    g0[1] = lds_byte_addr;                             // lds_addr  (bits 63:32)
    g0[2] = (unsigned)(ga & 0xFFFFFFFFu);              // global_addr[31:0]
    g0[3] = (unsigned)((ga >> 32) & 0x1FFFFFFu)        // global_addr[56:32]
          | (2u << 30);                                // type = 2 ("image")
    i32x8 g1;
    g1[0] = (1 << 16);                                 // wg_mask=0, data_size=1 (2B)
    g1[1] = (int)(((unsigned)cols & 0xFFFFu) << 16);   // tensor_dim0[15:0]
    g1[2] = (int)(((unsigned)cols >> 16) & 0xFFFFu)    // tensor_dim0[31:16]
          | (1 << 16);                                 // tensor_dim1 = 1
    g1[3] = (int)(((unsigned)cols & 0xFFFFu) << 16);   // tile_dim0 = cols
    g1[4] = 1;                                         // tile_dim1 = 1
    g1[5] = cols;                                      // tensor_dim0_stride lo
    g1[6] = 0;
    g1[7] = 0;
    const i32x4 z4 = {0, 0, 0, 0};
#if defined(__clang_major__) && __clang_major__ <= 22
    __builtin_amdgcn_tensor_load_to_lds(g0, g1, z4, z4, 0);
#else
    const i32x8 z8 = {0, 0, 0, 0, 0, 0, 0, 0};
    __builtin_amdgcn_tensor_load_to_lds(g0, g1, z4, z4, z8, 0);
#endif
}

// ---------------------------------------------------------------------------
// FAST PATH matvec: y = act(Wh @ x + bias), Wh row-major f16 [rows x cols].
// One 16-row tile per wave, K consumed 32/WMMA, fp32 accumulate.
//   A (16-bit 16x32): lane<16 -> K {0..7,16..23}; lane>=16 -> {8..15,24..31}
//   B (16-bit 32x16): lane<16 -> K {0..7,16..23}; lane>=16 -> {8..15,24..31}
// x (f16, in ws) staged into LDS by the Tensor Data Mover (wave 0 issues).
// ---------------------------------------------------------------------------
__global__ void k_matvec_wmma_h(const _Float16* __restrict__ Wh,
                                const _Float16* __restrict__ x16,
                                const float* __restrict__ bias,
                                float* __restrict__ y,
                                _Float16* __restrict__ y16,
                                int rows, int cols, int act)
{
    __shared__ _Float16 s_x[2 * H];   // cols <= 2048

    const int tid  = threadIdx.x;
    const int wid  = tid >> 5;
    const int lane = tid & 31;

    if (wid == 0) {  // wave-uniform: TDM ignores EXEC, issue once per block
        tdm_load_row_to_lds(x16, (unsigned)(uintptr_t)(void*)&s_x[0], cols);
        __builtin_amdgcn_s_wait_tensorcnt(0);
    }
    __syncthreads();

    const int hi   = lane >> 4;
    const int m    = lane & 15;
    const int tile = blockIdx.x * 8 + wid;
    if (tile * 16 >= rows) return;    // wave-uniform (EXEC all-1 for WMMA)

    const size_t rbase = (size_t)(tile * 16 + m) * (size_t)cols;

    v8f acc = {};
    for (int k0 = 0; k0 < cols; k0 += 32) {
        const int kA = k0 + (hi ? 8 : 0);
        const v8h c0 = *(const v8h*)(Wh + rbase + kA);
        const v8h c1 = *(const v8h*)(Wh + rbase + kA + 16);
        v16h A;
#pragma unroll
        for (int e = 0; e < 8; ++e) { A[e] = c0[e]; A[8 + e] = c1[e]; }

        const _Float16* xs = s_x + k0 + (hi ? 8 : 0);
        v16h B;
#pragma unroll
        for (int e = 0; e < 16; ++e) B[e] = xs[e < 8 ? e : e + 8];

        acc = __builtin_amdgcn_wmma_f32_16x16x32_f16(
                  false, A, false, B, (short)0, acc, false, false);

        if (k0 + 256 < cols)
            __builtin_prefetch(Wh + rbase + k0 + 256, 0, 3);
    }

    if ((lane & 15) == 0) {
        const int mb = tile * 16 + hi * 8;
#pragma unroll
        for (int r = 0; r < 8; ++r) {
            float v = acc[r] + bias[mb + r];
            if (act) v = fmaxf(v, 0.f);
            y[mb + r] = v;
            if (y16) y16[mb + r] = (_Float16)v;
        }
    }
}

// ---------------------------------------------------------------------------
// FALLBACK matvec (fp32 weights, f16 convert in-flight) — used if ws is small.
// x (fp32) staged into LDS with per-lane async b128 copies, then converted.
// ---------------------------------------------------------------------------
__global__ void k_matvec_wmma(const float* __restrict__ W,
                              const float* __restrict__ xbase,
                              const int*   __restrict__ xtok, int xtokoff,
                              const float* __restrict__ bias,
                              float* __restrict__ y,
                              int rows, int cols, int act)
{
    __shared__ float    s_xf32[2 * H];
    __shared__ _Float16 s_x[2 * H];

    const float* x = xtok ? (xbase + (size_t)xtok[xtokoff] * (size_t)cols) : xbase;

    const int tid = threadIdx.x;
    const int chunks = cols >> 2;     // 16-byte fp32 chunks
    const unsigned lds_f32 = (unsigned)(uintptr_t)(void*)&s_xf32[0];
    for (int c = tid; c < chunks; c += blockDim.x)
        async_b128_to_lds(lds_f32 + c * 16, (const char*)x + c * 16);
    wait_asynccnt0();
    __syncthreads();
    for (int i = tid; i < cols; i += blockDim.x)
        s_x[i] = (_Float16)s_xf32[i];
    __syncthreads();

    const int wid  = tid >> 5;
    const int lane = tid & 31;
    const int hi   = lane >> 4;
    const int m    = lane & 15;
    const int tile = blockIdx.x * 8 + wid;
    if (tile * 16 >= rows) return;

    const size_t rbase = (size_t)(tile * 16 + m) * (size_t)cols;

    v8f acc = {};
    for (int k0 = 0; k0 < cols; k0 += 32) {
        const int kA = k0 + (hi ? 8 : 0);
        const float4* p0 = (const float4*)(W + rbase + kA);
        const float4* p1 = (const float4*)(W + rbase + kA + 16);
        float4 a0 = p0[0], a1 = p0[1];
        float4 b0 = p1[0], b1 = p1[1];

        v16h A;
        A[0]  = (_Float16)a0.x; A[1]  = (_Float16)a0.y;
        A[2]  = (_Float16)a0.z; A[3]  = (_Float16)a0.w;
        A[4]  = (_Float16)a1.x; A[5]  = (_Float16)a1.y;
        A[6]  = (_Float16)a1.z; A[7]  = (_Float16)a1.w;
        A[8]  = (_Float16)b0.x; A[9]  = (_Float16)b0.y;
        A[10] = (_Float16)b0.z; A[11] = (_Float16)b0.w;
        A[12] = (_Float16)b1.x; A[13] = (_Float16)b1.y;
        A[14] = (_Float16)b1.z; A[15] = (_Float16)b1.w;

        const _Float16* xs = s_x + k0 + (hi ? 8 : 0);
        v16h B;
#pragma unroll
        for (int e = 0; e < 16; ++e) B[e] = xs[e < 8 ? e : e + 8];

        acc = __builtin_amdgcn_wmma_f32_16x16x32_f16(
                  false, A, false, B, (short)0, acc, false, false);

        if (k0 + 128 < cols)
            __builtin_prefetch(W + rbase + k0 + 128, 0, 3);
    }

    if ((lane & 15) == 0) {
        const int mb = tile * 16 + hi * 8;
#pragma unroll
        for (int r = 0; r < 8; ++r) {
            float v = acc[r] + bias[mb + r];
            if (act) v = fmaxf(v, 0.f);
            y[mb + r] = v;
        }
    }
}

// ---------------------------------------------------------------------------
// fp32 -> f16 converters (weights once per launch; emb row per step).
// ---------------------------------------------------------------------------
__global__ void k_cvt(const float* __restrict__ s, _Float16* __restrict__ d,
                      size_t n)
{
    size_t i = (size_t)blockIdx.x * blockDim.x + threadIdx.x;
    const size_t stride = (size_t)gridDim.x * blockDim.x;
    for (; i < n; i += stride) d[i] = (_Float16)s[i];
}

__global__ void k_cvt_vec(const float* __restrict__ base,
                          const int* __restrict__ tok, int tokoff,
                          _Float16* __restrict__ d, int n)
{
    const float* x = tok ? base + (size_t)tok[tokoff] * (size_t)n : base;
    const int i = blockIdx.x * blockDim.x + threadIdx.x;
    if (i < n) d[i] = (_Float16)x[i];
}

// ---------------------------------------------------------------------------
// Attention (single WG of 1024): w = softmax(attn_W @ [emb;h] + b);
// ctx = w @ enc_outs; cat2 = [emb;ctx] (f32 + optional f16 shadow).
// ---------------------------------------------------------------------------
__global__ void k_attn(const float* __restrict__ dec_emb,
                       const int*   __restrict__ tokp,
                       const float* __restrict__ h,
                       const float* __restrict__ attn_W,
                       const float* __restrict__ attn_b,
                       const float* __restrict__ enc_outs,
                       float* __restrict__ cat2,
                       _Float16* __restrict__ cat2_16)
{
    __shared__ float s_cat[2 * H];
    __shared__ float s_w[ML + 1];

    const int tid = threadIdx.x;
    const float* emb = dec_emb + (size_t)(*tokp) * H;
    for (int i = tid; i < H; i += blockDim.x) {
        s_cat[i]     = emb[i];
        s_cat[H + i] = h[i];
    }
    __syncthreads();

    const int wid = tid >> 5, lane = tid & 31;
    for (int row = wid; row < ML + 1; row += (int)(blockDim.x >> 5)) {
        float acc = 0.f;
        const float* wr = attn_W + (size_t)row * (2 * H);
        for (int k = lane; k < 2 * H; k += 32) acc += wr[k] * s_cat[k];
        for (int off = 16; off; off >>= 1) acc += __shfl_xor(acc, off, 32);
        if (lane == 0) s_w[row] = acc + attn_b[row];
    }
    __syncthreads();

    if (tid == 0) {
        float mx = -3.4e38f;
        for (int j = 0; j <= ML; ++j) mx = fmaxf(mx, s_w[j]);
        float s = 0.f;
        for (int j = 0; j <= ML; ++j) { float e = expf(s_w[j] - mx); s_w[j] = e; s += e; }
        float inv = 1.f / s;
        for (int j = 0; j <= ML; ++j) s_w[j] *= inv;
    }
    __syncthreads();

    for (int c = tid; c < H; c += blockDim.x) {
        float acc = 0.f;
        for (int j = 0; j <= ML; ++j) acc += s_w[j] * enc_outs[(size_t)j * H + c];
        cat2[c]     = s_cat[c];
        cat2[H + c] = acc;
        if (cat2_16) {
            cat2_16[c]     = (_Float16)s_cat[c];
            cat2_16[H + c] = (_Float16)acc;
        }
    }
}

// ---------------------------------------------------------------------------
// GRU elementwise combine (PyTorch GRUCell math) + optional f16 shadow.
// ---------------------------------------------------------------------------
__global__ void k_gru(const float* __restrict__ gi, const float* __restrict__ gh,
                      const float* __restrict__ hin,
                      float* __restrict__ hout,
                      _Float16* __restrict__ hout16,
                      float* __restrict__ extra_copy)
{
    const int i = blockIdx.x * blockDim.x + threadIdx.x;
    if (i >= H) return;
    const float r = 1.f / (1.f + expf(-(gi[i]         + gh[i])));
    const float z = 1.f / (1.f + expf(-(gi[i + H]     + gh[i + H])));
    const float n = tanhf(gi[i + 2 * H] + r * gh[i + 2 * H]);
    const float ho = (1.f - z) * n + z * hin[i];
    hout[i] = ho;
    if (hout16) hout16[i] = (_Float16)ho;
    if (extra_copy) extra_copy[i] = ho;
}

// ---------------------------------------------------------------------------
// log_softmax + argmax over V logits; writes output row and next token.
// ---------------------------------------------------------------------------
__global__ void k_logsoftmax_argmax(const float* __restrict__ logits,
                                    float* __restrict__ out_row,
                                    int* __restrict__ tokp)
{
    __shared__ float s_v[1024];
    __shared__ int   s_i[1024];
    const int tid = threadIdx.x;

    float lm = -3.4e38f; int li = 0x7fffffff;
    for (int i = tid; i < V; i += 1024) {
        float v = logits[i];
        if (v > lm) { lm = v; li = i; }
    }
    s_v[tid] = lm; s_i[tid] = li;
    __syncthreads();
    for (int s = 512; s > 0; s >>= 1) {
        if (tid < s) {
            float ov = s_v[tid + s]; int oi = s_i[tid + s];
            if (ov > s_v[tid] || (ov == s_v[tid] && oi < s_i[tid])) {
                s_v[tid] = ov; s_i[tid] = oi;
            }
        }
        __syncthreads();
    }
    const float mx = s_v[0];
    const int amax = s_i[0];
    __syncthreads();

    float ls = 0.f;
    for (int i = tid; i < V; i += 1024) ls += expf(logits[i] - mx);
    s_v[tid] = ls;
    __syncthreads();
    for (int s = 512; s > 0; s >>= 1) {
        if (tid < s) s_v[tid] += s_v[tid + s];
        __syncthreads();
    }
    const float lse = logf(s_v[0]);

    for (int i = tid; i < V; i += 1024) out_row[i] = logits[i] - mx - lse;
    if (tid == 0) *tokp = amax;
}

__global__ void k_init(float* __restrict__ hA, _Float16* __restrict__ h16A,
                       float* __restrict__ enc_pad_row, int* __restrict__ tokp)
{
    const int i = blockIdx.x * blockDim.x + threadIdx.x;
    if (i < H) {
        hA[i] = 0.f;
        enc_pad_row[i] = 0.f;
        if (h16A) h16A[i] = (_Float16)0.f;
    }
    if (i == 0) *tokp = SOS;
}

// ---------------------------------------------------------------------------
extern "C" void kernel_launch(void* const* d_in, const int* in_sizes, int n_in,
                              void* d_out, int out_size, void* d_ws, size_t ws_size,
                              hipStream_t stream)
{
    (void)in_sizes; (void)n_in; (void)out_size;

    const int*   tokens  = (const int*)  d_in[0];
    const float* enc_emb = (const float*)d_in[2];
    const float* enc_Wih = (const float*)d_in[3];
    const float* enc_Whh = (const float*)d_in[4];
    const float* enc_bih = (const float*)d_in[5];
    const float* enc_bhh = (const float*)d_in[6];
    const float* dec_emb = (const float*)d_in[7];
    const float* attn_W  = (const float*)d_in[8];
    const float* attn_b  = (const float*)d_in[9];
    const float* comb_W  = (const float*)d_in[10];
    const float* comb_b  = (const float*)d_in[11];
    const float* dec_Wih = (const float*)d_in[12];
    const float* dec_Whh = (const float*)d_in[13];
    const float* dec_bih = (const float*)d_in[14];
    const float* dec_bhh = (const float*)d_in[15];
    const float* out_W   = (const float*)d_in[16];
    const float* out_b   = (const float*)d_in[17];
    float* out = (float*)d_out;

    // ---- fp32 scratch ----
    float* ws       = (float*)d_ws;
    float* hA       = ws;                       // H
    float* hB       = hA + H;                   // H
    float* enc_outs = hB + H;                   // (ML+1)*H
    float* gi       = enc_outs + (ML + 1) * H;  // 3H
    float* gh       = gi + 3 * H;               // 3H
    float* cat2     = gh + 3 * H;               // 2H
    float* comb     = cat2 + 2 * H;             // H
    float* logits   = comb + H;                 // V
    int*   tokp     = (int*)(logits + V);

    // ---- f16 region (weights converted once per launch) ----
    size_t base_bytes = (size_t)((char*)(tokp + 1) - (char*)d_ws);
    size_t f16_off = (base_bytes + 255) & ~(size_t)255;
    _Float16* f16p    = (_Float16*)((char*)d_ws + f16_off);
    _Float16* x16     = f16p;  f16p += 2 * H;
    _Float16* h16A    = f16p;  f16p += H;
    _Float16* h16B    = f16p;  f16p += H;
    _Float16* comb16  = f16p;  f16p += H;
    _Float16* cat216  = f16p;  f16p += 2 * H;
    _Float16* encWih_h = f16p; f16p += (size_t)3 * H * H;
    _Float16* encWhh_h = f16p; f16p += (size_t)3 * H * H;
    _Float16* decWih_h = f16p; f16p += (size_t)3 * H * H;
    _Float16* decWhh_h = f16p; f16p += (size_t)3 * H * H;
    _Float16* combW_h  = f16p; f16p += (size_t)2 * H * H;
    _Float16* outW_h   = f16p; f16p += (size_t)V * H;
    const size_t need = (size_t)((char*)f16p - (char*)d_ws);
    const bool fast = ws_size >= need;

    if (fast) {
        k_cvt<<<2048, 256, 0, stream>>>(enc_Wih, encWih_h, (size_t)3 * H * H);
        k_cvt<<<2048, 256, 0, stream>>>(enc_Whh, encWhh_h, (size_t)3 * H * H);
        k_cvt<<<2048, 256, 0, stream>>>(dec_Wih, decWih_h, (size_t)3 * H * H);
        k_cvt<<<2048, 256, 0, stream>>>(dec_Whh, decWhh_h, (size_t)3 * H * H);
        k_cvt<<<2048, 256, 0, stream>>>(comb_W,  combW_h,  (size_t)2 * H * H);
        k_cvt<<<2048, 256, 0, stream>>>(out_W,   outW_h,   (size_t)V * H);
    }

    k_init<<<8, 256, 0, stream>>>(hA, fast ? h16A : nullptr,
                                  enc_outs + (size_t)ML * H, tokp);

    // ---------------- encoder ----------------
    for (int t = 0; t < ML; ++t) {
        const float* hin  = (t & 1) ? hB : hA;
        float*       hout = (t & 1) ? hA : hB;
        if (fast) {
            _Float16* h16in  = (t & 1) ? h16B : h16A;
            _Float16* h16out = (t & 1) ? h16A : h16B;
            k_cvt_vec<<<(H + 255) / 256, 256, 0, stream>>>(enc_emb, tokens, t, x16, H);
            k_matvec_wmma_h<<<24, 256, 0, stream>>>(encWih_h, x16, enc_bih,
                                                    gi, nullptr, 3 * H, H, 0);
            k_matvec_wmma_h<<<24, 256, 0, stream>>>(encWhh_h, h16in, enc_bhh,
                                                    gh, nullptr, 3 * H, H, 0);
            k_gru<<<(H + 255) / 256, 256, 0, stream>>>(gi, gh, hin, hout, h16out,
                                                       enc_outs + (size_t)t * H);
        } else {
            k_matvec_wmma<<<24, 256, 0, stream>>>(enc_Wih, enc_emb, tokens, t,
                                                  enc_bih, gi, 3 * H, H, 0);
            k_matvec_wmma<<<24, 256, 0, stream>>>(enc_Whh, hin, nullptr, 0,
                                                  enc_bhh, gh, 3 * H, H, 0);
            k_gru<<<(H + 255) / 256, 256, 0, stream>>>(gi, gh, hin, hout, nullptr,
                                                       enc_outs + (size_t)t * H);
        }
    }
    // after 128 steps the encoder hidden state lives in hA / h16A

    // ---------------- decoder ----------------
    for (int t = 0; t < ML; ++t) {
        const float* hin  = (t & 1) ? hB : hA;
        float*       hout = (t & 1) ? hA : hB;
        if (fast) {
            _Float16* h16in  = (t & 1) ? h16B : h16A;
            _Float16* h16out = (t & 1) ? h16A : h16B;
            k_attn<<<1, 1024, 0, stream>>>(dec_emb, tokp, hin, attn_W, attn_b,
                                           enc_outs, cat2, cat216);
            k_matvec_wmma_h<<<8, 256, 0, stream>>>(combW_h, cat216, comb_b,
                                                   comb, comb16, H, 2 * H, 1);
            k_matvec_wmma_h<<<24, 256, 0, stream>>>(decWih_h, comb16, dec_bih,
                                                    gi, nullptr, 3 * H, H, 0);
            k_matvec_wmma_h<<<24, 256, 0, stream>>>(decWhh_h, h16in, dec_bhh,
                                                    gh, nullptr, 3 * H, H, 0);
            k_gru<<<(H + 255) / 256, 256, 0, stream>>>(gi, gh, hin, hout, h16out,
                                                       nullptr);
            k_matvec_wmma_h<<<250, 256, 0, stream>>>(outW_h, h16out, out_b,
                                                     logits, nullptr, V, H, 0);
        } else {
            k_attn<<<1, 1024, 0, stream>>>(dec_emb, tokp, hin, attn_W, attn_b,
                                           enc_outs, cat2, nullptr);
            k_matvec_wmma<<<8, 256, 0, stream>>>(comb_W, cat2, nullptr, 0,
                                                 comb_b, comb, H, 2 * H, 1);
            k_matvec_wmma<<<24, 256, 0, stream>>>(dec_Wih, comb, nullptr, 0,
                                                  dec_bih, gi, 3 * H, H, 0);
            k_matvec_wmma<<<24, 256, 0, stream>>>(dec_Whh, hin, nullptr, 0,
                                                  dec_bhh, gh, 3 * H, H, 0);
            k_gru<<<(H + 255) / 256, 256, 0, stream>>>(gi, gh, hin, hout, nullptr,
                                                       nullptr);
            k_matvec_wmma<<<250, 256, 0, stream>>>(out_W, hout, nullptr, 0,
                                                   out_b, logits, V, H, 0);
        }
        k_logsoftmax_argmax<<<1, 1024, 0, stream>>>(logits, out + (size_t)t * V,
                                                    tokp);
    }
}
